// AdditiveLoRAAdapter_71442486002106
// MI455X (gfx1250) — matrix operand order repaired and checked by reading, not compiled, not run on hardware
//
#include <hip/hip_runtime.h>

#define D_IN   2048
#define D_OUT  2048
#define RANK   16
#define NEXP   16
#define RHID   64
#define NTOK   16384
#define ACOLS  (NEXP * RANK)   // 256
#define ALPHA  1.0f

typedef __attribute__((ext_vector_type(16))) __bf16 bf16x16;
typedef __attribute__((ext_vector_type(8)))  __bf16 bf16x8;
typedef __attribute__((ext_vector_type(4)))  __bf16 bf16x4;
typedef __attribute__((ext_vector_type(8)))  float  f32x8;

#define BM   128
#define BN   128
#define BK   32
#define ROWP 40   // padded LDS row: 32 halfs data + 8 pad = 80B (conflict-free b128 frags)

// out[n][m] = act( sum_k catX[n][k]*catW[m][k] + bias[m] )
// catX = [Xa (K1) | Xb (K2)], catW = [Wa | Wb]; bf16x3 emulation of f32 GEMM.
template<bool CAT, bool SILU, bool BIAS>
__global__ __launch_bounds__(256)
void gemm_bf16x3(const float* __restrict__ Xa, const float* __restrict__ Wa,
                 const float* __restrict__ Xb, const float* __restrict__ Wb,
                 const float* __restrict__ bias, float* __restrict__ out,
                 int M, int K1, int K2, int ldo)
{
    __shared__ __bf16 sXh[BM][ROWP], sXl[BM][ROWP];
    __shared__ __bf16 sWh[BN][ROWP], sWl[BN][ROWP];

    const int tid  = threadIdx.x;
    const int lane = tid & 31;
    const int wid  = tid >> 5;
    const int wm   = wid >> 2;      // 0..1  -> 64 rows
    const int wn   = wid & 3;       // 0..3  -> 32 cols
    const int lr   = lane & 15;
    const int kh   = lane >> 4;     // lane half selector
    const int n0   = blockIdx.x * BM;
    const int m0   = blockIdx.y * BN;

    f32x8 acc[4][2];
#pragma unroll
    for (int i = 0; i < 4; ++i)
#pragma unroll
        for (int j = 0; j < 2; ++j)
            acc[i][j] = (f32x8){0.f,0.f,0.f,0.f,0.f,0.f,0.f,0.f};

    const int nk = (K1 + K2) / BK;
    float4 rx[4], rw[4];

    auto g2r = [&](int kc) {
        const int k0 = kc * BK;
#pragma unroll
        for (int i = 0; i < 4; ++i) {
            int li  = i * 256 + tid;
            int row = li >> 3;           // 8 float4 per 32-float row
            int kq  = (li & 7) << 2;
            int gk  = k0 + kq;
            // X side (tokens, always in range)
            {
                const float* p;
                if (!CAT || gk < K1) p = Xa + (size_t)(n0 + row) * K1 + gk;
                else                 p = Xb + (size_t)(n0 + row) * K2 + (gk - K1);
                rx[i] = *(const float4*)p;
            }
            // W side (guard rows >= M)
            {
                int mrow = m0 + row;
                if (mrow < M) {
                    const float* p;
                    if (!CAT || gk < K1) p = Wa + (size_t)mrow * K1 + gk;
                    else                 p = Wb + (size_t)mrow * K2 + (gk - K1);
                    rw[i] = *(const float4*)p;
                } else {
                    rw[i] = make_float4(0.f, 0.f, 0.f, 0.f);
                }
            }
        }
    };

    auto r2lds = [&]() {
#pragma unroll
        for (int i = 0; i < 4; ++i) {
            int li  = i * 256 + tid;
            int row = li >> 3;
            int kq  = (li & 7) << 2;
            {
                float4 v = rx[i];
                __bf16 h0 = (__bf16)v.x, h1 = (__bf16)v.y, h2 = (__bf16)v.z, h3 = (__bf16)v.w;
                bf16x4 hh = {h0, h1, h2, h3};
                bf16x4 ll = {(__bf16)(v.x - (float)h0), (__bf16)(v.y - (float)h1),
                             (__bf16)(v.z - (float)h2), (__bf16)(v.w - (float)h3)};
                *(bf16x4*)&sXh[row][kq] = hh;
                *(bf16x4*)&sXl[row][kq] = ll;
            }
            {
                float4 v = rw[i];
                __bf16 h0 = (__bf16)v.x, h1 = (__bf16)v.y, h2 = (__bf16)v.z, h3 = (__bf16)v.w;
                bf16x4 hh = {h0, h1, h2, h3};
                bf16x4 ll = {(__bf16)(v.x - (float)h0), (__bf16)(v.y - (float)h1),
                             (__bf16)(v.z - (float)h2), (__bf16)(v.w - (float)h3)};
                *(bf16x4*)&sWh[row][kq] = hh;
                *(bf16x4*)&sWl[row][kq] = ll;
            }
        }
    };

    // A-frag (ISA 16-bit A 16x32): elems 0-7 -> K = kh*8..+7 ; elems 8-15 -> K = 16+kh*8..+7
    auto frA = [&](const __bf16 (*t)[ROWP], int rbase) -> bf16x16 {
        const __bf16* p = &t[rbase + lr][kh * 8];
        bf16x16 f;
        ((bf16x8*)&f)[0] = *(const bf16x8*)p;
        ((bf16x8*)&f)[1] = *(const bf16x8*)(p + 16);
        return f;
    };
    // B-frag (32x16, col = lane&15): elems 0-7 -> K = kh*8..+7 ; elems 8-15 -> K = 16+kh*8..+7
    auto frB = [&](const __bf16 (*t)[ROWP], int rbase) -> bf16x16 {
        const __bf16* p = &t[rbase + lr][kh * 8];
        bf16x16 f;
        ((bf16x8*)&f)[0] = *(const bf16x8*)p;
        ((bf16x8*)&f)[1] = *(const bf16x8*)(p + 16);
        return f;
    };

    g2r(0);
    for (int kc = 0; kc < nk; ++kc) {
        if (kc) __syncthreads();          // previous compute done before overwrite
        r2lds();
        __syncthreads();                  // tile visible to all waves
        if (kc + 1 < nk) g2r(kc + 1);     // prefetch next chunk while computing

        bf16x16 ah[4], al[4], bh[2], bl[2];
#pragma unroll
        for (int ms = 0; ms < 4; ++ms) {
            ah[ms] = frA(sXh, wm * 64 + ms * 16);
            al[ms] = frA(sXl, wm * 64 + ms * 16);
        }
#pragma unroll
        for (int ns = 0; ns < 2; ++ns) {
            bh[ns] = frB(sWh, wn * 32 + ns * 16);
            bl[ns] = frB(sWl, wn * 32 + ns * 16);
        }
#pragma unroll
        for (int ms = 0; ms < 4; ++ms)
#pragma unroll
            for (int ns = 0; ns < 2; ++ns) {
                f32x8 c = acc[ms][ns];
                c = __builtin_amdgcn_wmma_f32_16x16x32_bf16(false, al[ms], false, bh[ns], (short)0, c, false, false);
                c = __builtin_amdgcn_wmma_f32_16x16x32_bf16(false, ah[ms], false, bl[ns], (short)0, c, false, false);
                c = __builtin_amdgcn_wmma_f32_16x16x32_bf16(false, ah[ms], false, bh[ns], (short)0, c, false, false);
                acc[ms][ns] = c;
            }
    }

    // Epilogue: C layout -> row = base + kh*8 + g, col = base + lr
#pragma unroll
    for (int ms = 0; ms < 4; ++ms)
#pragma unroll
        for (int ns = 0; ns < 2; ++ns) {
            int col = m0 + wn * 32 + ns * 16 + lr;
            if (col < M) {
                float bv = BIAS ? bias[col] : 0.f;
#pragma unroll
                for (int g = 0; g < 8; ++g) {
                    int rown = n0 + wm * 64 + ms * 16 + kh * 8 + g;
                    float v = acc[ms][ns][g] + bv;
                    if (SILU) v = v / (1.f + __expf(-v));
                    out[(size_t)rown * ldo + col] = v;
                }
            }
        }
}

// logits = h @ rw2^T + rb2 + gates (first 16 experts), top-2 + softmax
__global__ __launch_bounds__(256)
void router_finish(const float* __restrict__ h, const float* __restrict__ rw2,
                   const float* __restrict__ rb2, const float* __restrict__ gates,
                   int* __restrict__ tope, float* __restrict__ topw)
{
    __shared__ float sw2[NEXP * RHID];
    __shared__ float sb[NEXP];
    for (int i = threadIdx.x; i < NEXP * RHID; i += 256) sw2[i] = rw2[i];
    if (threadIdx.x < NEXP) sb[threadIdx.x] = rb2[threadIdx.x] + gates[threadIdx.x];
    __syncthreads();

    int t = blockIdx.x * 256 + threadIdx.x;
    if (t >= NTOK) return;
    const float* hp = h + (size_t)t * RHID;
    float hr[RHID];
#pragma unroll
    for (int k = 0; k < RHID; k += 4) {
        float4 v = *(const float4*)(hp + k);
        hr[k] = v.x; hr[k + 1] = v.y; hr[k + 2] = v.z; hr[k + 3] = v.w;
    }
    float best = -1e30f, sec = -1e30f;
    int bi = 0, si = 0;
#pragma unroll
    for (int e = 0; e < NEXP; ++e) {
        float a = sb[e];
#pragma unroll
        for (int k = 0; k < RHID; ++k) a += hr[k] * sw2[e * RHID + k];
        if (a > best)      { sec = best; si = bi; best = a; bi = e; }
        else if (a > sec)  { sec = a; si = e; }
    }
    float e1  = __expf(sec - best);
    float inv = 1.f / (1.f + e1);
    tope[2 * t] = bi; tope[2 * t + 1] = si;
    topw[2 * t] = inv; topw[2 * t + 1] = e1 * inv;
}

// wxa[n][e*16+r] = ALPHA * coeff[n][e] * xa[n][e*16+r]   (in place)
__global__ __launch_bounds__(256)
void mask_wxa(float* __restrict__ xa, const int* __restrict__ tope,
              const float* __restrict__ topw)
{
    int i = blockIdx.x * 256 + threadIdx.x;
    if (i >= NTOK * ACOLS) return;
    int n = i >> 8, c = i & 255, e = c >> 4;
    int e0 = tope[2 * n], e1 = tope[2 * n + 1];
    float s = (e == e0) ? topw[2 * n] : (e == e1) ? topw[2 * n + 1] : 0.f;
    xa[i] *= ALPHA * s;
}

// Bt[o][e*16+r] = B[e][o][r]
__global__ __launch_bounds__(256)
void repack_B(const float* __restrict__ B, float* __restrict__ Bt)
{
    int i = blockIdx.x * 256 + threadIdx.x;
    if (i >= D_OUT * ACOLS) return;
    int o = i >> 8, c = i & 255, e = c >> 4, r = c & 15;
    Bt[i] = B[((size_t)e * D_OUT + o) * RANK + r];
}

extern "C" void kernel_launch(void* const* d_in, const int* in_sizes, int n_in,
                              void* d_out, int out_size, void* d_ws, size_t ws_size,
                              hipStream_t stream)
{
    const float* x     = (const float*)d_in[0];
    const float* W     = (const float*)d_in[1];
    const float* bias  = (const float*)d_in[2];
    const float* rw1   = (const float*)d_in[3];
    const float* rb1   = (const float*)d_in[4];
    const float* rw2   = (const float*)d_in[5];
    const float* rb2   = (const float*)d_in[6];
    const float* A     = (const float*)d_in[7];   // [16][16][2048] == [256][2048]
    const float* B     = (const float*)d_in[8];   // [16][2048][16]
    const float* gates = (const float*)d_in[9];
    float* out = (float*)d_out;

    float* h    = (float*)d_ws;                       // [NTOK][64]
    float* xa   = h    + (size_t)NTOK * RHID;         // [NTOK][256] (becomes wxa)
    float* Bt   = xa   + (size_t)NTOK * ACOLS;        // [2048][256]
    int*   tope = (int*)(Bt + (size_t)D_OUT * ACOLS); // [NTOK][2]
    float* topw = (float*)(tope + 2 * NTOK);          // [NTOK][2]

    dim3 blk(256);

    // 1) router hidden: h = silu(x @ rw1^T + rb1)   (M=64, padded to BN=128)
    gemm_bf16x3<false, true, true><<<dim3(NTOK / BM, 1), blk, 0, stream>>>(
        x, rw1, nullptr, nullptr, rb1, h, RHID, D_IN, 0, RHID);

    // 2) dense LoRA up: xa = x @ A_flat^T            (M=256)
    gemm_bf16x3<false, false, false><<<dim3(NTOK / BM, ACOLS / BN), blk, 0, stream>>>(
        x, A, nullptr, nullptr, nullptr, xa, ACOLS, D_IN, 0, ACOLS);

    // 3) repack B -> Bt[2048][256]
    repack_B<<<dim3((D_OUT * ACOLS) / 256), blk, 0, stream>>>(B, Bt);

    // 4) router logits + top-2 + softmax
    router_finish<<<dim3(NTOK / 256), blk, 0, stream>>>(h, rw2, rb2, gates, tope, topw);

    // 5) mask xa by router coeffs (in place -> wxa)
    mask_wxa<<<dim3((NTOK * ACOLS) / 256), blk, 0, stream>>>(xa, tope, topw);

    // 6) fused concat-K GEMM: out = [x | wxa] @ [W | Bt]^T + bias
    gemm_bf16x3<true, false, true><<<dim3(NTOK / BM, D_OUT / BN), blk, 0, stream>>>(
        x, W, xa, Bt, bias, out, D_OUT, D_IN, ACOLS, D_OUT);
}
